// NeighborEmbedding_61332132987236
// MI455X (gfx1250) — matrix operand order repaired and checked by reading, not compiled, not run on hardware
//
#include <hip/hip_runtime.h>

typedef __attribute__((ext_vector_type(16))) __bf16 v16bf;
typedef __attribute__((ext_vector_type(8)))  __bf16 v8bf;
typedef __attribute__((ext_vector_type(8)))  float  v8f;

#define BN_EPS 1e-5f
#define SLOPE 0.01f
#define NPTS (8 * 4096)     // B*N
#define NPERB 4096
#define KNN 32

// Native HW convert (emits v_cvt_*bf16* on gfx1250) instead of software RNE.
__device__ inline __bf16 f2bf(float f) { return (__bf16)f; }
__device__ inline uint2 pack4bf(float a, float b, float c, float d) {
    union { __bf16 h[4]; uint2 u; } v;
    v.h[0] = (__bf16)a; v.h[1] = (__bf16)b; v.h[2] = (__bf16)c; v.h[3] = (__bf16)d;
    return v.u;
}
// LeakyReLU(slope<1) == max(y, slope*y): mul+max, no cmp/cndmask.
__device__ inline float lrelu(float y) { return fmaxf(y, SLOPE * y); }

// ---------------------------------------------------------------------------
// Kernel 1: embed  h = LBRD2(LBRD1(x)) : [B,N,3] -> [B,N,64], plus |h|^2
// ---------------------------------------------------------------------------
__global__ __launch_bounds__(256) void embed_kernel(
    const float* __restrict__ x,
    const float* __restrict__ w1, const float* __restrict__ g1, const float* __restrict__ b1,
    const float* __restrict__ rm1, const float* __restrict__ rv1,
    const float* __restrict__ w2, const float* __restrict__ g2, const float* __restrict__ b2,
    const float* __restrict__ rm2, const float* __restrict__ rv2,
    float* __restrict__ hout, float* __restrict__ sqout)
{
    __shared__ float sw1[64 * 3];
    __shared__ float sw2[64 * 64];
    __shared__ float sa1[64], sb1[64], sa2[64], sb2[64];
    const int tid = threadIdx.x;
    for (int i = tid; i < 64 * 3; i += 256) sw1[i] = w1[i];
    for (int i = tid; i < 64 * 64; i += 256) sw2[i] = w2[i];
    for (int i = tid; i < 64; i += 256) {
        float a = g1[i] * rsqrtf(rv1[i] + BN_EPS); sa1[i] = a; sb1[i] = b1[i] - rm1[i] * a;
        float c = g2[i] * rsqrtf(rv2[i] + BN_EPS); sa2[i] = c; sb2[i] = b2[i] - rm2[i] * c;
    }
    __syncthreads();

    const int p = blockIdx.x * 256 + tid;
    if (p >= NPTS) return;
    const float x0 = x[p * 3 + 0], x1 = x[p * 3 + 1], x2 = x[p * 3 + 2];

    float h1[64];
#pragma unroll
    for (int c = 0; c < 64; ++c) {
        float s = sw1[c * 3] * x0 + sw1[c * 3 + 1] * x1 + sw1[c * 3 + 2] * x2;
        h1[c] = lrelu(s * sa1[c] + sb1[c]);
    }
    float sqacc = 0.f;
    float* hp = hout + (size_t)p * 64;
    for (int c2 = 0; c2 < 64; ++c2) {
        float s = 0.f;
#pragma unroll
        for (int c = 0; c < 64; ++c) s += h1[c] * sw2[c2 * 64 + c];
        s = lrelu(s * sa2[c2] + sb2[c2]);
        hp[c2] = s;
        sqacc += s * s;
    }
    sqout[p] = sqacc;
}

// ---------------------------------------------------------------------------
// Kernel 2: kNN — one wave per row, distances in per-wave 16KB LDS stripe,
// 32 extract-min passes with wave32 shuffles. Stores ABSOLUTE point indices.
// ---------------------------------------------------------------------------
__global__ __launch_bounds__(128) void knn_kernel(
    const float* __restrict__ h, const float* __restrict__ sq, int* __restrict__ idxo)
{
    __shared__ __align__(16) float sdist[4][NPERB];
    __shared__ __align__(16) float srow[4][64];
    const int wave = threadIdx.x >> 5;
    const int lid  = threadIdx.x & 31;
    const int row  = blockIdx.x * 4 + wave;
    const int b    = row >> 12;
    const int n0   = b << 12;                 // batch base
    const float* base = h + (size_t)n0 * 64;

    srow[wave][lid]      = h[(size_t)row * 64 + lid];
    srow[wave][lid + 32] = h[(size_t)row * 64 + lid + 32];
    const float sqc = sq[row];
    asm volatile("s_wait_dscnt 0" ::: "memory");

    float* dw = sdist[wave];
    for (int c = lid; c < NPERB; c += 32) {
        const float* cp = base + (size_t)c * 64;
        float dot = 0.f;
#pragma unroll
        for (int ci = 0; ci < 64; ci += 4) {
            float4 cv = *(const float4*)(cp + ci);
            float4 rv = *(const float4*)(&srow[wave][ci]);
            dot += cv.x * rv.x + cv.y * rv.y + cv.z * rv.z + cv.w * rv.w;
        }
        dw[c] = sqc + sq[n0 + c] - 2.f * dot;
    }
    asm volatile("s_wait_dscnt 0" ::: "memory");

    int* op = idxo + (size_t)row * KNN;
    for (int t = 0; t < KNN; ++t) {
        float mv = 3.4e38f; int mi = 0x7fffffff;
        for (int c = lid; c < NPERB; c += 32) {
            float v = dw[c];
            if (v < mv || (v == mv && c < mi)) { mv = v; mi = c; }
        }
#pragma unroll
        for (int off = 16; off > 0; off >>= 1) {
            float ov = __shfl_xor(mv, off, 32);
            int   oi = __shfl_xor(mi, off, 32);
            if (ov < mv || (ov == mv && oi < mi)) { mv = ov; mi = oi; }
        }
        if (lid == 0) { op[t] = n0 + mi; dw[mi] = 3.4e38f; }
        asm volatile("s_wait_dscnt 0" ::: "memory");
    }
}

// ---------------------------------------------------------------------------
// WMMA operand loaders (bf16, 16x16x32), contiguous k order (k = 16*(lane/16) + s) in both A and B.
// A (16x32, MxK): lanes 0-15 row=lane, K=0..15;
//                 lanes 16-31 row=lane-16, K=16..31.
// B (32x16, KxN): lanes 0-15 col=lane, K=0..15; lanes 16-31 col=lane-16, K=16..31.
// Matrices stored row-major in LDS with stride 128 bf16 (A: rows; B-source W: [co][ci]).
// ---------------------------------------------------------------------------
__device__ inline v16bf load_a_tile(const __bf16* M, int mt, int ks, int lid) {
    const int r  = mt * 16 + (lid & 15);
    const int k0 = ks * 32 + ((lid & 16) ? 16 : 0);
    union { v16bf v; v8bf h[2]; } u;
    u.h[0] = *(const v8bf*)(M + r * 128 + k0);
    u.h[1] = *(const v8bf*)(M + r * 128 + k0 + 8);
    return u.v;
}
__device__ inline v16bf load_b_tile(const __bf16* W, int nt, int ks, int lid) {
    const int n = nt * 16 + (lid & 15);
    const int k = ks * 32 + ((lid & 16) ? 16 : 0);
    return *(const v16bf*)(W + n * 128 + k);
}

// ---------------------------------------------------------------------------
// Kernel 3: fused edge-conv. Per point (one wave): gather 32 neighbors ->
// E[32x128] bf16 in LDS; Y1 = LReLU(BN(E @ W3^T)) via WMMA; out = max_k
// LReLU(BN(Y1 @ W4^T)). Never materializes the 537MB edge tensor.
// ---------------------------------------------------------------------------
__global__ __launch_bounds__(128) void edgeconv_kernel(
    const float* __restrict__ h, const int* __restrict__ idx,
    const float* __restrict__ w3, const float* __restrict__ g3, const float* __restrict__ b3,
    const float* __restrict__ rm3, const float* __restrict__ rv3,
    const float* __restrict__ w4, const float* __restrict__ g4, const float* __restrict__ b4,
    const float* __restrict__ rm4, const float* __restrict__ rv4,
    float* __restrict__ out)
{
    __shared__ __align__(32) __bf16 sW3[128 * 128];
    __shared__ __align__(32) __bf16 sW4[128 * 128];
    __shared__ __align__(32) __bf16 sE[4][32 * 128];
    __shared__ __align__(32) __bf16 sY[4][32 * 128];
    __shared__ float sA3[128], sB3[128], sA4[128], sB4[128];

    const int tid = threadIdx.x;
    for (int i = tid; i < 128 * 128; i += 128) { sW3[i] = f2bf(w3[i]); sW4[i] = f2bf(w4[i]); }
    for (int i = tid; i < 128; i += 128) {
        float a = g3[i] * rsqrtf(rv3[i] + BN_EPS); sA3[i] = a; sB3[i] = b3[i] - rm3[i] * a;
        float c = g4[i] * rsqrtf(rv4[i] + BN_EPS); sA4[i] = c; sB4[i] = b4[i] - rm4[i] * c;
    }
    __syncthreads();

    const int wave = tid >> 5;
    const int lid  = tid & 31;
    __bf16* E = sE[wave];
    __bf16* Y = sY[wave];
    const int nwaves = gridDim.x * 4;
    const int gwave  = blockIdx.x * 4 + wave;

    for (int p = gwave; p < NPTS; p += nwaves) {
        // ---- gather: lane == neighbor slot kk (wave32 => 32 lanes == k) ----
        const float* ctr = h + (size_t)p * 64;
        const int    j   = idx[(size_t)p * KNN + lid];       // absolute point id
        const float* nf  = h + (size_t)j * 64;
#pragma unroll
        for (int ci = 0; ci < 64; ci += 4) {
            float4 cv = *(const float4*)(ctr + ci);
            float4 nv = *(const float4*)(nf + ci);
            *(uint2*)&E[lid * 128 + ci]      = pack4bf(nv.x - cv.x, nv.y - cv.y,
                                                       nv.z - cv.z, nv.w - cv.w);
            *(uint2*)&E[lid * 128 + 64 + ci] = pack4bf(cv.x, cv.y, cv.z, cv.w);
        }
        asm volatile("s_wait_dscnt 0" ::: "memory");

        // ---- layer 1: Y = LReLU(BN(E[32x128] @ W3^T)) ----
        for (int mt = 0; mt < 2; ++mt) {
            for (int nt = 0; nt < 8; ++nt) {
                v8f acc = {};
#pragma unroll
                for (int ks = 0; ks < 4; ++ks) {
                    v16bf a = load_a_tile(E, mt, ks, lid);
                    v16bf bb = load_b_tile(sW3, nt, ks, lid);
                    acc = __builtin_amdgcn_wmma_f32_16x16x32_bf16(
                        false, a, false, bb, (short)0, acc, false, false);
                }
                const int   co = nt * 16 + (lid & 15);
                const float aa = sA3[co], bo = sB3[co];
                const int   mbase = mt * 16 + ((lid & 16) ? 8 : 0);
#pragma unroll
                for (int i = 0; i < 8; ++i)
                    Y[(mbase + i) * 128 + co] = f2bf(lrelu(acc[i] * aa + bo));
            }
        }
        asm volatile("s_wait_dscnt 0" ::: "memory");

        // ---- layer 2 + max over k: out = max_m LReLU(BN(Y @ W4^T)) ----
        float* outp = out + (size_t)p * 128;
        for (int nt = 0; nt < 8; ++nt) {
            float cmax = -3.4e38f;
            for (int mt = 0; mt < 2; ++mt) {
                v8f acc = {};
#pragma unroll
                for (int ks = 0; ks < 4; ++ks) {
                    v16bf a = load_a_tile(Y, mt, ks, lid);
                    v16bf bb = load_b_tile(sW4, nt, ks, lid);
                    acc = __builtin_amdgcn_wmma_f32_16x16x32_bf16(
                        false, a, false, bb, (short)0, acc, false, false);
                }
                const int   co = nt * 16 + (lid & 15);
                const float aa = sA4[co], bo = sB4[co];
#pragma unroll
                for (int i = 0; i < 8; ++i)
                    cmax = fmaxf(cmax, lrelu(acc[i] * aa + bo));
            }
            cmax = fmaxf(cmax, __shfl_xor(cmax, 16, 32));
            if (lid < 16) outp[nt * 16 + lid] = cmax;
        }
        asm volatile("s_wait_dscnt 0" ::: "memory");
    }
}

// ---------------------------------------------------------------------------
extern "C" void kernel_launch(void* const* d_in, const int* in_sizes, int n_in,
                              void* d_out, int out_size, void* d_ws, size_t ws_size,
                              hipStream_t stream) {
    const float* x   = (const float*)d_in[0];
    const float* w1  = (const float*)d_in[1];
    const float* g1  = (const float*)d_in[2];
    const float* b1  = (const float*)d_in[3];
    const float* rm1 = (const float*)d_in[4];
    const float* rv1 = (const float*)d_in[5];
    const float* w2  = (const float*)d_in[6];
    const float* g2  = (const float*)d_in[7];
    const float* b2  = (const float*)d_in[8];
    const float* rm2 = (const float*)d_in[9];
    const float* rv2 = (const float*)d_in[10];
    const float* w3  = (const float*)d_in[11];
    const float* g3  = (const float*)d_in[12];
    const float* b3  = (const float*)d_in[13];
    const float* rm3 = (const float*)d_in[14];
    const float* rv3 = (const float*)d_in[15];
    const float* w4  = (const float*)d_in[16];
    const float* g4  = (const float*)d_in[17];
    const float* b4  = (const float*)d_in[18];
    const float* rm4 = (const float*)d_in[19];
    const float* rv4 = (const float*)d_in[20];
    float* out = (float*)d_out;

    // workspace carve: h [NPTS*64] f32, sq [NPTS] f32, idx [NPTS*32] i32 (~12.2 MB)
    float* h  = (float*)d_ws;
    float* sq = h + (size_t)NPTS * 64;
    int*   ix = (int*)(sq + NPTS);

    embed_kernel<<<NPTS / 256, 256, 0, stream>>>(x, w1, g1, b1, rm1, rv1,
                                                 w2, g2, b2, rm2, rv2, h, sq);
    knn_kernel<<<NPTS / 4, 128, 0, stream>>>(h, sq, ix);
    edgeconv_kernel<<<512, 128, 0, stream>>>(h, ix, w3, g3, b3, rm3, rv3,
                                             w4, g4, b4, rm4, rv4, out);
}
